// SparseAttentionAdapter_29712583754354
// MI455X (gfx1250) — matrix operand order repaired and checked by reading, not compiled, not run on hardware
//
#include <hip/hip_runtime.h>

// ---------------------------------------------------------------------------
// CDNA5 (gfx1250) NSA forward. wave32; bf16-native WMMA GEMMs with 2x-unrolled
// ping-pong double buffering (no register-copy rotation), WMMA attention
// tiles, VALU gather branch for fine attention.
// ---------------------------------------------------------------------------

typedef __bf16 bf16_t;
typedef __attribute__((ext_vector_type(16))) __bf16 v16bf;
typedef __attribute__((ext_vector_type(8)))  __bf16 v8bf;
typedef __attribute__((ext_vector_type(8)))  float  v8f;

#define SCALE_DH 0.08838834764831845f   /* 128^-0.5 */
#define NEGINF   (-1e30f)

__device__ __forceinline__ v8f wmma_bf16(v16bf a, v16bf b, v8f c) {
  return __builtin_amdgcn_wmma_f32_16x16x32_bf16(false, a, false, b, (short)0, c,
                                                 false, false);
}

// ---- bf16-native fragment loads (16B-aligned vector loads, no conversion) --
// A layout: lanes 0-15 -> M=lane, K {kc..kc+15};
//           lanes 16-31 -> M=lane-16, K {kc+16..kc+31}.
__device__ __forceinline__ v16bf load_a_bf(const bf16_t* A, int lda, int row0,
                                           int kc, int lane) {
  int m   = row0 + (lane & 15);
  int klo = (lane & 16) ? 16 : 0;
  const bf16_t* p = A + (long)m * lda + kc + klo;
  v8bf lo = *(const v8bf*)p;
  v8bf hi = *(const v8bf*)(p + 8);
  return __builtin_shufflevector(lo, hi, 0, 1, 2, 3, 4, 5, 6, 7,
                                 8, 9, 10, 11, 12, 13, 14, 15);
}

// B fragment for C = A @ W^T (B[k][n] = W[n][k]); W row-major [N,K] bf16.
__device__ __forceinline__ v16bf load_bT_bf(const bf16_t* W, int ldw, int col0,
                                            int kc, int lane) {
  int n  = col0 + (lane & 15);
  int kb = (lane & 16) ? 16 : 0;
  const bf16_t* p = W + (long)n * ldw + kc + kb;
  v8bf lo = *(const v8bf*)p;
  v8bf hi = *(const v8bf*)(p + 8);
  return __builtin_shufflevector(lo, hi, 0, 1, 2, 3, 4, 5, 6, 7,
                                 8, 9, 10, 11, 12, 13, 14, 15);
}

// ---- fp32-source fragment loads (attention kernels, LDS-staged P) ----------
__device__ __forceinline__ v16bf load_a_frag(const float* A, int lda, int row0,
                                             int kc, int lane) {
  int m   = row0 + (lane & 15);
  int klo = (lane & 16) ? 16 : 0;
  const float* p = A + (long)m * lda + kc + klo;
  v16bf f;
#pragma unroll
  for (int i = 0; i < 8; ++i) {
    f[i]     = (bf16_t)p[i];
    f[8 + i] = (bf16_t)p[8 + i];
  }
  return f;
}

// B fragment for C = P @ V (natural orientation); V row-major [J, ldv], guard k<jmax.
__device__ __forceinline__ v16bf load_b_rows(const float* V, int ldv, int col0,
                                             int kc, int jmax, int lane) {
  int n  = col0 + (lane & 15);
  int kb = (lane & 16) ? 16 : 0;
  v16bf f;
#pragma unroll
  for (int t = 0; t < 16; ++t) {
    int k = kc + kb + t;
    float x = (k < jmax) ? V[(long)k * ldv + n] : 0.f;
    f[t] = (bf16_t)x;
  }
  return f;
}

// ---------------------------------------------------------------------------
// fp32 -> bf16 conversion (pairs).
// ---------------------------------------------------------------------------
__global__ void conv_bf16_kernel(const float* __restrict__ src,
                                 bf16_t* __restrict__ dst, long n2) {
  long i = (long)blockIdx.x * 256 + threadIdx.x;
  if (i < n2) {
    float2 v = ((const float2*)src)[i];
    dst[2 * i]     = (bf16_t)v.x;
    dst[2 * i + 1] = (bf16_t)v.y;
  }
}

// ---------------------------------------------------------------------------
// Tiled GEMM, bf16 operands: out = act(A @ W^T + bias). 4 waves / block,
// each wave computes 16x64 (one A fragment reused across 4 B tiles).
// K-loop unrolled 2x with two disjoint fragment buffer sets (ping-pong):
// loads for one buffer issue while the other buffer's WMMAs execute, and no
// register-copy rotation is needed. Requires K % 64 == 0 (true here).
// Optional fp32 (C) and bf16 (Cbf) outputs.
// remap: m -> (m/64)*65+1+(m%64)  (compressed-KV slot layout).
// ---------------------------------------------------------------------------
__global__ __launch_bounds__(128) void gemm_bf16w(
    const bf16_t* __restrict__ A, const bf16_t* __restrict__ W,
    const float* __restrict__ bias, float* __restrict__ C,
    bf16_t* __restrict__ Cbf, int M, int N, int K, int remap, int act) {
  int lane = threadIdx.x & 31;
  int wave = threadIdx.x >> 5;
  int col0 = blockIdx.x * 64;
  int row0 = blockIdx.y * 64 + wave * 16;
  v8f zero = {0.f, 0.f, 0.f, 0.f, 0.f, 0.f, 0.f, 0.f};
  v8f acc[4];
#pragma unroll
  for (int nt = 0; nt < 4; ++nt) acc[nt] = zero;

  // --- preload pair-0 first chunk into buffer set 0 ---
  v16bf a0 = load_a_bf(A, K, row0, 0, lane);
  v16bf b0[4];
#pragma unroll
  for (int nt = 0; nt < 4; ++nt)
    b0[nt] = load_bT_bf(W, K, col0 + nt * 16, 0, lane);

  int npairs = K >> 6;          // K % 64 == 0
  int k0 = 0;
  for (int p = 0; p + 1 < npairs; ++p, k0 += 64) {
    // load chunk k0+32 into set 1, consume set 0
    v16bf a1 = load_a_bf(A, K, row0, k0 + 32, lane);
    v16bf b1[4];
#pragma unroll
    for (int nt = 0; nt < 4; ++nt)
      b1[nt] = load_bT_bf(W, K, col0 + nt * 16, k0 + 32, lane);
#pragma unroll
    for (int nt = 0; nt < 4; ++nt)
      acc[nt] = wmma_bf16(a0, b0[nt], acc[nt]);
    // load chunk k0+64 into set 0, consume set 1
    a0 = load_a_bf(A, K, row0, k0 + 64, lane);
#pragma unroll
    for (int nt = 0; nt < 4; ++nt)
      b0[nt] = load_bT_bf(W, K, col0 + nt * 16, k0 + 64, lane);
#pragma unroll
    for (int nt = 0; nt < 4; ++nt)
      acc[nt] = wmma_bf16(a1, b1[nt], acc[nt]);
  }
  // --- drain final pair (chunks k0, k0+32) ---
  {
    v16bf a1 = load_a_bf(A, K, row0, k0 + 32, lane);
    v16bf b1[4];
#pragma unroll
    for (int nt = 0; nt < 4; ++nt)
      b1[nt] = load_bT_bf(W, K, col0 + nt * 16, k0 + 32, lane);
#pragma unroll
    for (int nt = 0; nt < 4; ++nt)
      acc[nt] = wmma_bf16(a0, b0[nt], acc[nt]);
#pragma unroll
    for (int nt = 0; nt < 4; ++nt)
      acc[nt] = wmma_bf16(a1, b1[nt], acc[nt]);
  }

  int mb = row0 + ((lane & 16) ? 8 : 0);
#pragma unroll
  for (int nt = 0; nt < 4; ++nt) {
    int n = col0 + nt * 16 + (lane & 15);
    float bv = bias ? bias[n] : 0.f;
#pragma unroll
    for (int r = 0; r < 8; ++r) {
      float x = acc[nt][r] + bv;
      if (act == 1) x = fmaxf(x, 0.f);
      long m  = mb + r;
      long dr = remap ? ((m >> 6) * 65 + 1 + (m & 63)) : m;
      if (C)   C[dr * N + n]   = x;
      if (Cbf) Cbf[dr * N + n] = (bf16_t)x;
    }
  }
}

// ---------------------------------------------------------------------------
// Build compress-MLP inputs (bf16): ck_in = k + k_pos, cv_in = v + v_pos; mem token.
// ---------------------------------------------------------------------------
__global__ void build_cin_kernel(const float* __restrict__ qkv,
                                 const float* __restrict__ k_pos,
                                 const float* __restrict__ v_pos,
                                 const float* __restrict__ mem_kv,
                                 bf16_t* __restrict__ ckinh, bf16_t* __restrict__ cvinh,
                                 float* __restrict__ ck, float* __restrict__ cv) {
  long i = (long)blockIdx.x * 256 + threadIdx.x;
  const long N = 8L * 64 * 2048;
  if (i < N) {
    int d = i & 127; long r = i >> 7;
    int t = r & 15;  r >>= 4;
    int c = r & 63;  r >>= 6;
    int h = (int)r;
    int s = c * 16 + t;
    long q   = (long)s * 6144 + h * 128 + d;
    long pos = (long)h * 2048 + t * 128 + d;
    ckinh[i] = (bf16_t)(qkv[q + 4096] + k_pos[pos]);
    cvinh[i] = (bf16_t)(qkv[q + 5120] + v_pos[pos]);
  }
  if (i < 8 * 128) {
    int h = (int)(i >> 7), d = (int)(i & 127);
    ck[(long)h * 65 * 128 + d] = mem_kv[(long)h * 128 + d];
    cv[(long)h * 65 * 128 + d] = mem_kv[8L * 128 + (long)h * 128 + d];
  }
}

// ---------------------------------------------------------------------------
// Coarse attention: wave per (head, 16-query tile). 65 keys padded to 80/96.
// ---------------------------------------------------------------------------
__global__ __launch_bounds__(32) void coarse_attn_kernel(
    const float* __restrict__ qkv, const float* __restrict__ ck,
    const float* __restrict__ cv, float* __restrict__ csim,
    float* __restrict__ cout_) {
  __shared__ float sc[16][96];
  int lane = threadIdx.x;
  int h = blockIdx.y, kvh = h >> 2;
  int row0 = blockIdx.x * 16;
  const float* Q  = qkv + h * 128;                 // row stride 6144
  const float* Kc = ck + (long)kvh * 65 * 128;
  const float* Vc = cv + (long)kvh * 65 * 128;

  v8f zero = {0.f, 0.f, 0.f, 0.f, 0.f, 0.f, 0.f, 0.f};
  v8f acc[5];
#pragma unroll
  for (int jt = 0; jt < 5; ++jt) acc[jt] = zero;

  for (int kc = 0; kc < 128; kc += 32) {
    v16bf a = load_a_frag(Q, 6144, row0, kc, lane);
#pragma unroll
    for (int jt = 0; jt < 5; ++jt) {
      int key = jt * 16 + (lane & 15);
      int kb  = (lane & 16) ? 16 : 0;
      const float* kp = Kc + (long)(key < 65 ? key : 0) * 128 + kc + kb;
      v16bf b;
#pragma unroll
      for (int t = 0; t < 16; ++t) b[t] = (bf16_t)(key < 65 ? kp[t] : 0.f);
      acc[jt] = wmma_bf16(a, b, acc[jt]);
    }
  }

  int nl = lane & 15;
  int mb = (lane & 16) ? 8 : 0;
#pragma unroll
  for (int jt = 0; jt < 5; ++jt) {
#pragma unroll
    for (int r = 0; r < 8; ++r) {
      int j = jt * 16 + nl;
      int m = mb + r, s = row0 + m;
      float val = acc[jt][r] * SCALE_DH;
      bool vis = (j < 65) && (j == 0 || j * 16 <= s);   // ck_seq[j] < s
      if (!vis) val = NEGINF;
      sc[m][j] = val;
      if (j < 65) csim[((long)h * 1024 + s) * 65 + j] = val;
    }
  }
  __syncthreads();
  if (lane < 16) {
    int m = lane;
    float mx = NEGINF;
    for (int j = 0; j < 65; ++j) mx = fmaxf(mx, sc[m][j]);
    float sum = 0.f;
    for (int j = 0; j < 65; ++j) { float e = __expf(sc[m][j] - mx); sc[m][j] = e; sum += e; }
    float inv = 1.f / sum;
    for (int j = 0; j < 65; ++j) sc[m][j] *= inv;
    for (int j = 65; j < 96; ++j) sc[m][j] = 0.f;
  }
  __syncthreads();

#pragma unroll
  for (int nt = 0; nt < 8; ++nt) {
    v8f o = zero;
    for (int kc = 0; kc < 96; kc += 32) {
      v16bf a = load_a_frag(&sc[0][0], 96, 0, kc, lane);
      v16bf b = load_b_rows(Vc, 128, nt * 16, kc, 65, lane);
      o = wmma_bf16(a, b, o);
    }
    int d = nt * 16 + nl;
#pragma unroll
    for (int r = 0; r < 8; ++r)
      cout_[((long)h * 1024 + row0 + mb + r) * 128 + d] = o[r];
  }
}

// ---------------------------------------------------------------------------
// Importance softmax + top-4 block selection per (kvh, s).
// ---------------------------------------------------------------------------
__global__ void topk_kernel(const float* __restrict__ csim,
                            int* __restrict__ selidx, int* __restrict__ selok) {
  int idx = blockIdx.x * 64 + threadIdx.x;
  if (idx >= 8 * 1024) return;
  int kvh = idx >> 10, s = idx & 1023;
  float imp[64];
  for (int j = 0; j < 64; ++j) {
    float sum = 0.f;
    for (int g = 0; g < 4; ++g)
      sum += csim[(((long)(kvh * 4 + g) * 1024) + s) * 65 + 1 + j];
    imp[j] = sum * 0.25f;
  }
  float mx = -1e3f;                                  // sentinel
  for (int j = 0; j < 64; ++j) mx = fmaxf(mx, imp[j]);
  float sum = __expf(-1e3f - mx);
  for (int j = 0; j < 64; ++j) { float e = __expf(imp[j] - mx); imp[j] = e; sum += e; }
  float inv = 1.f / sum;
  for (int j = 0; j < 64; ++j) imp[j] *= inv;

  int* si = selidx + (long)idx * 5;
  int* so = selok  + (long)idx * 5;
  for (int k = 0; k < 4; ++k) {
    float best = -1.f; int bi = 0;
    for (int j = 0; j < 64; ++j)
      if (imp[j] > best) { best = imp[j]; bi = j; }
    si[k] = bi;
    so[k] = (best > 1e-10f) ? 1 : 0;
    imp[bi] = -2.f;
  }
  si[4] = s >> 4;                                    // own block
  so[4] = 1;
}

// ---------------------------------------------------------------------------
// Interleaved RoPE (theta = 10000) on q (32 heads) and k (8 kv heads).
// ---------------------------------------------------------------------------
__global__ void rope_kernel(const float* __restrict__ qkv,
                            float* __restrict__ rq, float* __restrict__ rk) {
  long i = (long)blockIdx.x * 256 + threadIdx.x;
  const long NQ = 32L * 1024 * 64, NK = 8L * 1024 * 64;
  if (i >= NQ + NK) return;
  bool isq = i < NQ;
  long r = isq ? i : i - NQ;
  int pp = r & 63; r >>= 6;
  int s  = r & 1023; r >>= 10;
  int h  = (int)r;
  const float* src = qkv + (long)s * 6144 + (isq ? h * 128 : 4096 + h * 128) + 2 * pp;
  float inv = __expf(-(2.f * pp / 128.f) * 9.210340371976184f);  // 10000^-(2p/128)
  float ang = (float)s * inv;
  float sn, cs;
  __sincosf(ang, &sn, &cs);
  float x0 = src[0], x1 = src[1];
  float* dst = (isq ? rq : rk) + ((long)h * 1024 + s) * 128 + 2 * pp;
  dst[0] = x0 * cs - x1 * sn;
  dst[1] = x1 * cs + x0 * sn;
}

// ---------------------------------------------------------------------------
// Fine attention: wave per (query s, head h). 5 blocks * 16 = 80 gathered keys.
// ---------------------------------------------------------------------------
__global__ __launch_bounds__(32) void fine_attn_kernel(
    const float* __restrict__ rq, const float* __restrict__ rk,
    const float* __restrict__ qkv, const int* __restrict__ selidx,
    const int* __restrict__ selok, float* __restrict__ fout) {
  __shared__ float qs[128];
  __shared__ float p[80];
  __shared__ int   krow[80];
  int lane = threadIdx.x;
  int s = blockIdx.x, h = blockIdx.y, kvh = h >> 2;
  const float* q = rq + ((long)h * 1024 + s) * 128;
#pragma unroll
  for (int i = 0; i < 4; ++i) qs[lane + 32 * i] = q[lane + 32 * i];
  __syncthreads();
  const int* sel = selidx + ((long)kvh * 1024 + s) * 5;
  const int* ok  = selok  + ((long)kvh * 1024 + s) * 5;
#pragma unroll
  for (int t = 0; t < 3; ++t) {
    int j = lane + 32 * t;
    if (j < 80) {
      int blk = j >> 4, tt = j & 15;
      int b = sel[blk];
      bool good = ok[blk] && (blk < 4 || tt <= (s & 15));
      int kr = b * 16 + tt;
      krow[j] = kr;
      const float* kp = rk + ((long)kvh * 1024 + kr) * 128;
      float d = 0.f;
      for (int e = 0; e < 128; ++e) d += qs[e] * kp[e];
      p[j] = good ? d * SCALE_DH : NEGINF;
    }
  }
  __syncthreads();
  float lm = NEGINF;
#pragma unroll
  for (int t = 0; t < 3; ++t) { int j = lane + 32 * t; if (j < 80) lm = fmaxf(lm, p[j]); }
  for (int o = 16; o; o >>= 1) lm = fmaxf(lm, __shfl_xor(lm, o, 32));
  float ls = 0.f;
#pragma unroll
  for (int t = 0; t < 3; ++t) { int j = lane + 32 * t; if (j < 80) ls += __expf(p[j] - lm); }
  for (int o = 16; o; o >>= 1) ls += __shfl_xor(ls, o, 32);
  float inv = 1.f / ls;
#pragma unroll
  for (int t = 0; t < 3; ++t) { int j = lane + 32 * t; if (j < 80) p[j] = __expf(p[j] - lm) * inv; }
  __syncthreads();
  int d0 = lane * 4;
  float o0 = 0.f, o1 = 0.f, o2 = 0.f, o3 = 0.f;
  for (int j = 0; j < 80; ++j) {
    float pj = p[j];
    const float* vp = qkv + (long)krow[j] * 6144 + 5120 + kvh * 128 + d0;
    o0 += pj * vp[0]; o1 += pj * vp[1]; o2 += pj * vp[2]; o3 += pj * vp[3];
  }
  float* dst = fout + ((long)h * 1024 + s) * 128 + d0;
  dst[0] = o0; dst[1] = o1; dst[2] = o2; dst[3] = o3;
}

// ---------------------------------------------------------------------------
// Sliding window (prev + own 64-window): wave per (head, 16-query tile).
// ---------------------------------------------------------------------------
__global__ __launch_bounds__(32) void swin_attn_kernel(
    const float* __restrict__ rq, const float* __restrict__ rk,
    const float* __restrict__ qkv, float* __restrict__ sout) {
  __shared__ float sc[16][128];
  int lane = threadIdx.x;
  int h = blockIdx.y, kvh = h >> 2;
  int row0 = blockIdx.x * 16;
  int w = row0 >> 6;
  int kbase = w * 64 - 64;
  const float* Q  = rq + (long)h * 1024 * 128;
  const float* Kr = rk + (long)kvh * 1024 * 128;

  v8f zero = {0.f, 0.f, 0.f, 0.f, 0.f, 0.f, 0.f, 0.f};
  v8f acc[8];
#pragma unroll
  for (int jt = 0; jt < 8; ++jt) acc[jt] = zero;

  for (int kc = 0; kc < 128; kc += 32) {
    v16bf a = load_a_frag(Q, 128, row0, kc, lane);
#pragma unroll
    for (int jt = 0; jt < 8; ++jt) {
      int key = kbase + jt * 16 + (lane & 15);
      int kb  = (lane & 16) ? 16 : 0;
      const float* kp = Kr + (long)(key >= 0 ? key : 0) * 128 + kc + kb;
      v16bf b;
#pragma unroll
      for (int t = 0; t < 16; ++t) b[t] = (bf16_t)(key >= 0 ? kp[t] : 0.f);
      acc[jt] = wmma_bf16(a, b, acc[jt]);
    }
  }

  int nl = lane & 15;
  int mb = (lane & 16) ? 8 : 0;
#pragma unroll
  for (int jt = 0; jt < 8; ++jt) {
#pragma unroll
    for (int r = 0; r < 8; ++r) {
      int j = jt * 16 + nl;
      int kabs = kbase + j;
      int iabs = row0 + mb + r;
      int rel = iabs - kabs;
      float val = acc[jt][r] * SCALE_DH;
      if (!(kabs >= 0 && rel >= 0 && rel <= 64)) val = NEGINF;
      sc[mb + r][j] = val;
    }
  }
  __syncthreads();
  if (lane < 16) {
    int m = lane;
    float mx = NEGINF;
    for (int j = 0; j < 128; ++j) mx = fmaxf(mx, sc[m][j]);
    float sum = 0.f;
    for (int j = 0; j < 128; ++j) { float e = __expf(sc[m][j] - mx); sc[m][j] = e; sum += e; }
    float inv = 1.f / sum;
    for (int j = 0; j < 128; ++j) sc[m][j] *= inv;
  }
  __syncthreads();

#pragma unroll
  for (int nt = 0; nt < 8; ++nt) {
    v8f o = zero;
    for (int kc = 0; kc < 128; kc += 32) {
      v16bf a = load_a_frag(&sc[0][0], 128, 0, kc, lane);
      int n  = nt * 16 + nl;
      int kb = (lane & 16) ? 16 : 0;
      v16bf b;
#pragma unroll
      for (int t = 0; t < 16; ++t) {
        int row = kbase + kc + kb + t;
        b[t] = (bf16_t)qkv[(long)(row >= 0 ? row : 0) * 6144 + 5120 + kvh * 128 + n];
      }
      o = wmma_bf16(a, b, o);
    }
    int d = nt * 16 + nl;
#pragma unroll
    for (int r = 0; r < 8; ++r)
      sout[((long)h * 1024 + row0 + mb + r) * 128 + d] = o[r];
  }
}

// ---------------------------------------------------------------------------
// 3-way gate GEMV: gates[s][g] = sigmoid(X[s] . w_gate[g] + b_gate[g]).
// ---------------------------------------------------------------------------
__global__ __launch_bounds__(32) void gate_kernel(const float* __restrict__ X,
                                                  const float* __restrict__ wg,
                                                  const float* __restrict__ bg,
                                                  float* __restrict__ gates) {
  int lane = threadIdx.x;
  int g = blockIdx.x, s = blockIdx.y;
  const float* x  = X  + (long)s * 4096;
  const float* wp = wg + (long)g * 4096;
  float acc = 0.f;
  for (int t = 0; t < 128; ++t) acc += x[lane + 32 * t] * wp[lane + 32 * t];
  for (int o = 16; o; o >>= 1) acc += __shfl_xor(acc, o, 32);
  if (lane == 0) gates[(long)s * 96 + g] = 1.f / (1.f + __expf(-(acc + bg[g])));
}

__global__ void combine_kernel(const float* __restrict__ gates,
                               const float* __restrict__ cout_,
                               const float* __restrict__ fout,
                               const float* __restrict__ sout,
                               bf16_t* __restrict__ combh) {
  long i = (long)blockIdx.x * 256 + threadIdx.x;
  if (i >= 1024L * 4096) return;
  int s = (int)(i >> 12), cc = (int)(i & 4095);
  int h = cc >> 7, d = cc & 127;
  const float* g = gates + (long)s * 96 + h * 3;
  long src = ((long)h * 1024 + s) * 128 + d;
  combh[i] = (bf16_t)(g[0] * cout_[src] + g[1] * fout[src] + g[2] * sout[src]);
}

// ---------------------------------------------------------------------------
extern "C" void kernel_launch(void* const* d_in, const int* in_sizes, int n_in,
                              void* d_out, int out_size, void* d_ws, size_t ws_size,
                              hipStream_t stream) {
  const float* X      = (const float*)d_in[0];
  const float* w_qkv  = (const float*)d_in[1];
  const float* w_o    = (const float*)d_in[2];
  const float* k_pos  = (const float*)d_in[3];
  const float* v_pos  = (const float*)d_in[4];
  const float* k_w1   = (const float*)d_in[5];
  const float* k_b1   = (const float*)d_in[6];
  const float* k_w2   = (const float*)d_in[7];
  const float* k_b2   = (const float*)d_in[8];
  const float* v_w1   = (const float*)d_in[9];
  const float* v_b1   = (const float*)d_in[10];
  const float* v_w2   = (const float*)d_in[11];
  const float* v_b2   = (const float*)d_in[12];
  const float* mem_kv = (const float*)d_in[13];
  const float* w_gate = (const float*)d_in[14];
  const float* b_gate = (const float*)d_in[15];
  float* out = (float*)d_out;

  float* ws = (float*)d_ws;
  size_t off = 0;                               // float units; keep 16B-aligned
  float* qkv   = ws + off; off += 1024L * 6144;
  float* ck    = ws + off; off += 8L * 65 * 128;
  float* cv    = ws + off; off += 8L * 65 * 128;
  float* csim  = ws + off; off += 32L * 1024 * 65;
  float* cout_ = ws + off; off += 32L * 1024 * 128;
  float* rq    = ws + off; off += 32L * 1024 * 128;
  float* rk    = ws + off; off += 8L * 1024 * 128;
  float* fout  = ws + off; off += 32L * 1024 * 128;
  float* sout  = ws + off; off += 32L * 1024 * 128;
  float* gates = ws + off; off += 1024L * 96;
  int* selidx  = (int*)(ws + off); off += 8L * 1024 * 5;
  int* selok   = (int*)(ws + off); off += 8L * 1024 * 5;
  off = (off + 3) & ~3ul;                       // 16B align for bf16 vectors
  bf16_t* xh     = (bf16_t*)(ws + off); off += 1024L * 4096 / 2;
  bf16_t* wqkvh  = (bf16_t*)(ws + off); off += 6144L * 4096 / 2;
  bf16_t* woh    = (bf16_t*)(ws + off); off += 4096L * 4096 / 2;
  bf16_t* kw1h   = (bf16_t*)(ws + off); off += 2048L * 2048 / 2;
  bf16_t* vw1h   = (bf16_t*)(ws + off); off += 2048L * 2048 / 2;
  bf16_t* kw2h   = (bf16_t*)(ws + off); off += 128L * 2048 / 2;
  bf16_t* vw2h   = (bf16_t*)(ws + off); off += 128L * 2048 / 2;
  bf16_t* ckinh  = (bf16_t*)(ws + off); off += 512L * 2048 / 2;
  bf16_t* cvinh  = (bf16_t*)(ws + off); off += 512L * 2048 / 2;
  bf16_t* hkh    = (bf16_t*)(ws + off); off += 512L * 2048 / 2;
  bf16_t* hvh    = (bf16_t*)(ws + off); off += 512L * 2048 / 2;
  bf16_t* combh  = (bf16_t*)(ws + off); off += 1024L * 4096 / 2;

  dim3 wv(32);
  auto conv = [&](const float* s, bf16_t* d, long n) {
    long n2 = n / 2;
    conv_bf16_kernel<<<(int)((n2 + 255) / 256), 256, 0, stream>>>(s, d, n2);
  };
  // 0) bf16 operand conversion (weights + activations)
  conv(X, xh, 1024L * 4096);
  conv(w_qkv, wqkvh, 6144L * 4096);
  conv(w_o, woh, 4096L * 4096);
  conv(k_w1, kw1h, 2048L * 2048);
  conv(v_w1, vw1h, 2048L * 2048);
  conv(k_w2, kw2h, 128L * 2048);
  conv(v_w2, vw2h, 128L * 2048);

  // 1) QKV projection
  gemm_bf16w<<<dim3(6144 / 64, 1024 / 64), 128, 0, stream>>>(
      xh, wqkvh, nullptr, qkv, nullptr, 1024, 6144, 4096, 0, 0);
  // 2) compress-MLP inputs + mem token
  build_cin_kernel<<<(8 * 64 * 2048 + 255) / 256, 256, 0, stream>>>(
      qkv, k_pos, v_pos, mem_kv, ckinh, cvinh, ck, cv);
  // 3-6) compress MLPs (relu layer1 -> bf16, layer2 with slot remap -> fp32)
  gemm_bf16w<<<dim3(2048 / 64, 512 / 64), 128, 0, stream>>>(
      ckinh, kw1h, k_b1, nullptr, hkh, 512, 2048, 2048, 0, 1);
  gemm_bf16w<<<dim3(2048 / 64, 512 / 64), 128, 0, stream>>>(
      cvinh, vw1h, v_b1, nullptr, hvh, 512, 2048, 2048, 0, 1);
  gemm_bf16w<<<dim3(128 / 64, 512 / 64), 128, 0, stream>>>(
      hkh, kw2h, k_b2, ck, nullptr, 512, 128, 2048, 1, 0);
  gemm_bf16w<<<dim3(128 / 64, 512 / 64), 128, 0, stream>>>(
      hvh, vw2h, v_b2, cv, nullptr, 512, 128, 2048, 1, 0);
  // 7) coarse attention
  coarse_attn_kernel<<<dim3(64, 32), wv, 0, stream>>>(qkv, ck, cv, csim, cout_);
  // 8) importance + top-4
  topk_kernel<<<dim3(128), 64, 0, stream>>>(csim, selidx, selok);
  // 9) RoPE
  rope_kernel<<<(int)((32L * 1024 * 64 + 8L * 1024 * 64 + 255) / 256), 256, 0, stream>>>(
      qkv, rq, rk);
  // 10) fine attention
  fine_attn_kernel<<<dim3(1024, 32), wv, 0, stream>>>(rq, rk, qkv, selidx, selok, fout);
  // 11) sliding window
  swin_attn_kernel<<<dim3(64, 32), wv, 0, stream>>>(rq, rk, qkv, sout);
  // 12) gates + combine (combine emits bf16 for the final GEMM)
  gate_kernel<<<dim3(96, 1024), wv, 0, stream>>>(X, w_gate, b_gate, gates);
  combine_kernel<<<(int)((1024L * 4096 + 255) / 256), 256, 0, stream>>>(
      gates, cout_, fout, sout, combh);
  // 13) output projection
  gemm_bf16w<<<dim3(4096 / 64, 1024 / 64), 128, 0, stream>>>(
      combh, woh, nullptr, out, nullptr, 1024, 4096, 4096, 0, 0);
}
